// TTTLayer_59133109731575
// MI455X (gfx1250) — compile-verified, matrix operand order checked by reading, not run on hardware
//
#include <hip/hip_runtime.h>
#include <hip/hip_bf16.h>

// ---------------------------------------------------------------------------
// TTT layer, single GD step, fused into one WMMA kernel chain (gfx1250).
//   B=8, T=4096, H=1024, D=256  -> R = 32768 rows.
// Math (stability grad is 0 since state==orig_state at step 0):
//   Z  = S@W1 + b1 ; Hr = relu(Z)
//   E  = Hr@W2 + b2 - target
//   G1 = E@W2^T ; Gz = G1 * (Z>0)
//   out = S - lr*(2/(B*T*H)) * (Gz@W1^T)
// Precision: bf16 WMMA (v_wmma_f32_16x16x32_bf16) with f32 accumulate.
// Weight panels staged into LDS by the Tensor Data Mover (tensor_load_to_lds,
// TENSORcnt), double-buffered so the DMA overlaps the WMMA stream.
// ---------------------------------------------------------------------------

typedef __attribute__((ext_vector_type(16))) __bf16 v16bf;
typedef __attribute__((ext_vector_type(8)))  float  v8f;
typedef __attribute__((ext_vector_type(4)))  float  f32x4;
typedef __attribute__((ext_vector_type(4)))  unsigned int u32x4;
typedef __attribute__((ext_vector_type(8)))  int    i32x8;
typedef __attribute__((ext_vector_type(4)))  int    i32x4;

#define H_DIM 1024
#define D_DIM 256
#define ROWS  32768
#define WG_THREADS 256
#define ROWS_PER_WG 128   // 8 waves * 16 rows

#if __has_builtin(__builtin_amdgcn_tensor_load_to_lds)
#define USE_TDM 1
#else
#define USE_TDM 0
#endif

// Packed bf16 weight-fragment arrays in workspace (each frag = 32x16 = 512 bf16 = 1KB):
//   W1p  : K=H (32 kb), N=D (16 nt)  -> 512 frags
//   W2p  : K=D ( 8 kb), N=H (64 nt)  -> 512 frags
//   W2Tp : K=H (32 kb), N=D (16 nt)  -> 512 frags   (W2T[k][n] = W2[n][k])
//   W1Tp : K=D ( 8 kb), N=H (64 nt)  -> 512 frags   (W1T[k][n] = W1[n][k])
// frag index f = kb*Nt + nt; within frag: element = lane*16 + j,
// lane = (khalf<<4)|col, k_local = ((j>>3)<<4) | (khalf<<3) | (j&7)
// (runs-of-8 K interleave, mirroring the ISA 16-bit A-matrix layout).
#define WP_W1P  0
#define WP_W2P  262144
#define WP_W2TP 524288
#define WP_W1TP 786432
#define WP_TOTAL 1048576   // bf16 elems = 2 MB

// Element index inside a per-wave A-fragment-packed LDS region for
// row m (0..15) and K index d (0..K-1). Region holds K/32 frags of 512 elems.
__device__ __forceinline__ int frag_addr(int m, int d) {
  int kb    = d >> 5;
  int kl    = d & 31;
  int khalf = (kl >> 3) & 1;
  int j     = ((kl >> 4) << 3) | (kl & 7);
  int lane  = (khalf << 4) | m;
  return kb * 512 + lane * 16 + j;
}

#if USE_TDM
// Issue a TDM copy of one 16 KB weight panel (global -> LDS), 1-D tile.
// D# per CDNA5 ISA:
//   g0: [1:0]=count=1, [63:32]=lds_addr, [120:64]=global_addr, [127:126]=type=2
//   g1: data_size=3 (8B units), tensor_dim0=tile_dim0=tensor_dim0_stride=2048,
//       tensor_dim1=1, everything else 0 (no multicast, no pad, no iterate).
__device__ __forceinline__ void tdm_issue_panel(const __bf16* src,
                                                unsigned lds_byte_off) {
  unsigned long long ga = (unsigned long long)src;
  u32x4 g0;
  g0[0] = 1u;                                            // count=1 (valid D#)
  g0[1] = lds_byte_off;                                  // LDS dest address
  g0[2] = (unsigned)(ga & 0xFFFFFFFFull);                // global addr [31:0]
  g0[3] = (unsigned)((ga >> 32) & 0x01FFFFFFull)         // global addr [56:32]
          | (2u << 30);                                  // type = 2 ("image")
  i32x8 g1;
  g1[0] = 0x00030000;   // workgroup_mask=0, data_size=3 (8-byte units)
  g1[1] = 0x08000000;   // tensor_dim0[15:0]=2048 @ bits 63:48
  g1[2] = 0x00010000;   // tensor_dim1[15:0]=1    @ bits 95:80
  g1[3] = 0x08000000;   // tile_dim0=2048         @ bits 127:112
  g1[4] = 0;            // tile_dim1=0 (1-D tile), tile_dim2=0
  g1[5] = 0x00000800;   // tensor_dim0_stride[31:0]=2048 @ bits 191:160
  g1[6] = 0;
  g1[7] = 0;
  i32x4 gz = {0, 0, 0, 0};
#if defined(__clang_major__) && (__clang_major__ >= 23)
  i32x8 gz8 = {0, 0, 0, 0, 0, 0, 0, 0};
  __builtin_amdgcn_tensor_load_to_lds(g0, g1, gz, gz, gz8, 0);
#else
  __builtin_amdgcn_tensor_load_to_lds(g0, g1, gz, gz, 0);
#endif
}
#endif

// ---------------------------------------------------------------------------
// Prep: pack W1/W2 (+transposes) into bf16 WMMA B-fragment layout.
// ---------------------------------------------------------------------------
__global__ __launch_bounds__(256)
void ttt_prep_weights(const float* __restrict__ W1, const float* __restrict__ W2,
                      unsigned short* __restrict__ wp) {
  int gid    = blockIdx.x * 256 + threadIdx.x;      // 0 .. 1048575
  int arr    = gid >> 18;                           // which packed array
  int idx    = gid & 0x3FFFF;
  int frag   = idx >> 9;
  int within = idx & 511;
  int lane   = within >> 4;
  int j      = within & 15;
  int col    = lane & 15;
  int khalf  = lane >> 4;
  int klocal = ((j >> 3) << 4) | (khalf << 3) | (j & 7);
  float v;
  if (arr == 0) {            // W1p  [H x D]
    int kb = frag >> 4, nt = frag & 15;
    v = W1[(kb * 32 + klocal) * D_DIM + nt * 16 + col];
  } else if (arr == 1) {     // W2p  [D x H]
    int kb = frag >> 6, nt = frag & 63;
    v = W2[(kb * 32 + klocal) * H_DIM + nt * 16 + col];
  } else if (arr == 2) {     // W2Tp [H x D] = W2[n][k]
    int kb = frag >> 4, nt = frag & 15;
    v = W2[(nt * 16 + col) * H_DIM + kb * 32 + klocal];
  } else {                   // W1Tp [D x H] = W1[n][k]
    int kb = frag >> 6, nt = frag & 63;
    v = W1[(nt * 16 + col) * D_DIM + kb * 32 + klocal];
  }
  __bf16 bv = (__bf16)v;
  wp[gid] = __builtin_bit_cast(unsigned short, bv);
}

// ---------------------------------------------------------------------------
// Fused main kernel: 256 blocks x 256 threads (8 waves), 16 rows per wave.
// LDS: per-wave Hrelu frags (8KB), per-wave scratch frags for E/Gz (8KB),
//      2 x 16KB double-buffered weight-fragment panels (TDM-staged).
// ---------------------------------------------------------------------------
__global__ __launch_bounds__(WG_THREADS)
void ttt_fused(const float* __restrict__ S, const float* __restrict__ Tg,
               const float* __restrict__ b1, const float* __restrict__ b2,
               const float* __restrict__ stepp,
               const unsigned short* __restrict__ wp_u,
               float* __restrict__ out) {
  __shared__ __bf16 lds_hrel[8][4096];                  // 8 waves * 8KB
  __shared__ __bf16 lds_scr [8][4096];                  // 8 waves * 8KB
  __shared__ __align__(1024) __bf16 lds_panel[2][8192]; // 2 x 16KB panels

  const int tid     = threadIdx.x;
  const int wave    = tid >> 5;
  const int lane    = tid & 31;
  const int col16   = lane & 15;
  const int mhalf   = lane >> 4;                 // +8 rows for upper lane half
  const int rowbase = blockIdx.x * ROWS_PER_WG + wave * 16;

  const __bf16* wp   = reinterpret_cast<const __bf16*>(wp_u);
  const __bf16* W1p  = wp + WP_W1P;
  const __bf16* W2p  = wp + WP_W2P;
  const __bf16* W2Tp = wp + WP_W2TP;
  const __bf16* W1Tp = wp + WP_W1TP;

  const v8f vzero = {0.f, 0.f, 0.f, 0.f, 0.f, 0.f, 0.f, 0.f};
  const float coef = stepp[0] * (2.0f / 33554432.0f);   // lr * 2/(B*T*H)

#if USE_TDM
  const unsigned poff0 = (unsigned)(unsigned long long)&lds_panel[0][0];
  const unsigned poff1 = (unsigned)(unsigned long long)&lds_panel[1][0];
#endif

  // Stage panel (async TDM issue by wave 0; no-op in fallback build).
  auto panel_stage = [&](const __bf16* src, int buf) {
#if USE_TDM
    if (wave == 0) tdm_issue_panel(src, buf ? poff1 : poff0);
#else
    (void)src; (void)buf;
#endif
  };
  // Make panel `buf` consumable by all waves (TDM: wait TENSORcnt + barrier;
  // fallback: cooperative register copy between two barriers).
  auto panel_ready = [&](const __bf16* src, int buf) {
#if USE_TDM
    (void)src; (void)buf;
    if (wave == 0) __builtin_amdgcn_s_wait_tensorcnt(0);
    __syncthreads();
#else
    __syncthreads();
    const f32x4* g = (const f32x4*)src;
    f32x4*       p = (f32x4*)&lds_panel[buf][0];
#pragma unroll
    for (int i = 0; i < 4; ++i) p[tid * 4 + i] = g[tid * 4 + i];
    __syncthreads();
#endif
  };

  // =========================== Phase 1: Z = S@W1 ==========================
  v8f zacc[16];
#pragma unroll
  for (int t = 0; t < 16; ++t) zacc[t] = vzero;

  panel_stage(W1p, 0);
  for (int kb = 0; kb < 32; ++kb) {
    const int buf = kb & 1;
    panel_ready(W1p + kb * 8192, buf);
    if (kb + 1 < 32) panel_stage(W1p + (kb + 1) * 8192, buf ^ 1);
    // Build bf16 A-frag of S: lane holds row (lane&15), runs-of-8 K interleave.
    v16bf a;
    {
      const float* sp = S + (rowbase + col16) * H_DIM + kb * 32 + mhalf * 8;
      f32x4 f0 = *(const f32x4*)(sp);
      f32x4 f1 = *(const f32x4*)(sp + 4);
      f32x4 f2 = *(const f32x4*)(sp + 16);
      f32x4 f3 = *(const f32x4*)(sp + 20);
#pragma unroll
      for (int q = 0; q < 4; ++q) {
        a[q]      = (__bf16)f0[q];
        a[4 + q]  = (__bf16)f1[q];
        a[8 + q]  = (__bf16)f2[q];
        a[12 + q] = (__bf16)f3[q];
      }
    }
#pragma unroll
    for (int nt = 0; nt < 16; ++nt) {
      v16bf b = *(const v16bf*)(&lds_panel[buf][nt * 512 + lane * 16]);
      zacc[nt] = __builtin_amdgcn_wmma_f32_16x16x32_bf16(
          false, a, false, b, (short)0, zacc[nt], false, false);
    }
  }

  // relu + bias -> Hrelu A-frags in LDS; keep (Z>0) mask bit-packed (128 bits).
  unsigned mask[4] = {0u, 0u, 0u, 0u};
#pragma unroll
  for (int nt = 0; nt < 16; ++nt) {
    int d0 = nt * 16 + col16;
    float bb = b1[d0];
#pragma unroll
    for (int r = 0; r < 8; ++r) {
      int m = r + (mhalf << 3);
      float z = zacc[nt][r] + bb;
      if (z > 0.0f) mask[nt >> 2] |= 1u << (((nt & 3) << 3) + r);
      float h = z > 0.0f ? z : 0.0f;
      lds_hrel[wave][frag_addr(m, d0)] = (__bf16)h;
    }
  }

  // ============ Phase 2+3 fused over 4 chunks of 256 H-columns ============
  // P_chunk = Hr@W2 (+b2) ; E = P - T ; G1 += E_chunk @ W2T_chunkK
  v8f g1[16];
#pragma unroll
  for (int t = 0; t < 16; ++t) g1[t] = vzero;

  for (int nc = 0; nc < 4; ++nc) {
    v8f pacc[16];
#pragma unroll
    for (int t = 0; t < 16; ++t) pacc[t] = vzero;

    panel_stage(W2p + (nc * 16) * 512, 0);
    for (int kb = 0; kb < 8; ++kb) {                 // K = D = 256
      const int buf = kb & 1;
      panel_ready(W2p + (kb * 64 + nc * 16) * 512, buf);
      if (kb + 1 < 8) panel_stage(W2p + ((kb + 1) * 64 + nc * 16) * 512, buf ^ 1);
      v16bf a = *(const v16bf*)(&lds_hrel[wave][kb * 512 + lane * 16]);
#pragma unroll
      for (int nt = 0; nt < 16; ++nt) {
        v16bf b = *(const v16bf*)(&lds_panel[buf][nt * 512 + lane * 16]);
        pacc[nt] = __builtin_amdgcn_wmma_f32_16x16x32_bf16(
            false, a, false, b, (short)0, pacc[nt], false, false);
      }
    }

    // E = P + b2 - target  -> bf16 A-frags (K = local H index) in scratch LDS
#pragma unroll
    for (int nt = 0; nt < 16; ++nt) {
      int col = nc * 256 + nt * 16 + col16;
      float bb = b2[col];
#pragma unroll
      for (int r = 0; r < 8; ++r) {
        int m = r + (mhalf << 3);
        int row = rowbase + m;
        float e = pacc[nt][r] + bb - Tg[row * H_DIM + col];
        lds_scr[wave][frag_addr(m, nt * 16 + col16)] = (__bf16)e;
      }
    }

    panel_stage(W2Tp + (nc * 8) * 8192, 0);
    for (int kb = 0; kb < 8; ++kb) {                 // K chunk of H
      const int buf = kb & 1;
      panel_ready(W2Tp + (nc * 8 + kb) * 8192, buf);
      if (kb + 1 < 8) panel_stage(W2Tp + (nc * 8 + kb + 1) * 8192, buf ^ 1);
      v16bf a = *(const v16bf*)(&lds_scr[wave][kb * 512 + lane * 16]);
#pragma unroll
      for (int nt = 0; nt < 16; ++nt) {
        v16bf b = *(const v16bf*)(&lds_panel[buf][nt * 512 + lane * 16]);
        g1[nt] = __builtin_amdgcn_wmma_f32_16x16x32_bf16(
            false, a, false, b, (short)0, g1[nt], false, false);
      }
    }
  }

  // Gz = G1 * mask -> bf16 A-frags in scratch LDS (K = D = 256)
#pragma unroll
  for (int nt = 0; nt < 16; ++nt) {
#pragma unroll
    for (int r = 0; r < 8; ++r) {
      int m = r + (mhalf << 3);
      bool on = (mask[nt >> 2] >> (((nt & 3) << 3) + r)) & 1u;
      float gz = on ? g1[nt][r] : 0.0f;
      lds_scr[wave][frag_addr(m, nt * 16 + col16)] = (__bf16)gz;
    }
  }

  // ========== Phase 4: out = S - coef * (Gz @ W1T), 4 chunks of 256 =======
  for (int nc = 0; nc < 4; ++nc) {
    v8f gacc[16];
#pragma unroll
    for (int t = 0; t < 16; ++t) gacc[t] = vzero;

    panel_stage(W1Tp + (nc * 16) * 512, 0);
    for (int kb = 0; kb < 8; ++kb) {                 // K = D = 256
      const int buf = kb & 1;
      panel_ready(W1Tp + (kb * 64 + nc * 16) * 512, buf);
      if (kb + 1 < 8) panel_stage(W1Tp + ((kb + 1) * 64 + nc * 16) * 512, buf ^ 1);
      v16bf a = *(const v16bf*)(&lds_scr[wave][kb * 512 + lane * 16]);
#pragma unroll
      for (int nt = 0; nt < 16; ++nt) {
        v16bf b = *(const v16bf*)(&lds_panel[buf][nt * 512 + lane * 16]);
        gacc[nt] = __builtin_amdgcn_wmma_f32_16x16x32_bf16(
            false, a, false, b, (short)0, gacc[nt], false, false);
      }
    }

#pragma unroll
    for (int nt = 0; nt < 16; ++nt) {
      int col = nc * 256 + nt * 16 + col16;
#pragma unroll
      for (int r = 0; r < 8; ++r) {
        int m = r + (mhalf << 3);
        int off = (rowbase + m) * H_DIM + col;
        out[off] = S[off] - coef * gacc[nt][r];
      }
    }
  }
}

// ---------------------------------------------------------------------------
extern "C" void kernel_launch(void* const* d_in, const int* in_sizes, int n_in,
                              void* d_out, int out_size, void* d_ws, size_t ws_size,
                              hipStream_t stream) {
  const float* S  = (const float*)d_in[0];   // state  (8,4096,1024)
  const float* Tg = (const float*)d_in[1];   // target (8,4096,1024)
  const float* W1 = (const float*)d_in[2];   // (1024,256)
  const float* b1 = (const float*)d_in[3];   // (256,)
  const float* W2 = (const float*)d_in[4];   // (256,1024)
  const float* b2 = (const float*)d_in[5];   // (1024,)
  const float* lr = (const float*)d_in[6];   // scalar
  float* out = (float*)d_out;
  unsigned short* wp = (unsigned short*)d_ws;    // 2 MB packed bf16 weights

  ttt_prep_weights<<<WP_TOTAL / 256, 256, 0, stream>>>(W1, W2, wp);
  ttt_fused<<<ROWS / ROWS_PER_WG, WG_THREADS, 0, stream>>>(
      S, Tg, b1, b2, lr, wp, out);
}
